// TriMul_13022340841595
// MI455X (gfx1250) — matrix operand order repaired and checked by reading, not compiled, not run on hardware
//
#include <hip/hip_runtime.h>
#include <hip/hip_bf16.h>

// ---------------------------------------------------------------------------
// TriMul (triangle multiplicative update), B=1, L=512, D=H=128, MI455X/gfx1250
// All matrix contractions on v_wmma_f32_16x16x32_bf16 (wave32).
// Layouts chosen so WMMA A/B fragments are contiguous b128 loads from L2.
// ---------------------------------------------------------------------------

#define LDIM 512
#define LL   262144        // L*L
#define DDIM 128
#define HDIM 128
#define LN_EPS 1e-5f

typedef __attribute__((ext_vector_type(16))) __bf16 bf16x16;
typedef __attribute__((ext_vector_type(8)))  float  floatx8;

union ABFrag {               // 32 bytes: one 16-bit WMMA A or B fragment / lane
  uint4 q[2];
  unsigned short us[16];
  bf16x16 v;
};

__device__ __forceinline__ unsigned short f2bf(float f) {
  unsigned int u = __float_as_uint(f);
  u += 0x7FFFu + ((u >> 16) & 1u);          // round-to-nearest-even
  return (unsigned short)(u >> 16);
}
__device__ __forceinline__ float sigmoid_f(float x) {
  return 1.0f / (1.0f + __expf(-x));
}

// ---------------------------------------------------------------------------
// Kernel 0: convert 6 weight matrices (5 proj [H,D] + w_out [D,H]) to bf16.
// Order in wbf: left, right, lgate, rgate, ogate, wout (each 16384 elems).
// ---------------------------------------------------------------------------
__global__ void __launch_bounds__(256)
k0_convert_weights(const float* __restrict__ wl, const float* __restrict__ wr,
                   const float* __restrict__ wlg, const float* __restrict__ wrg,
                   const float* __restrict__ wog, const float* __restrict__ wout,
                   unsigned short* __restrict__ wbf) {
  int idx = blockIdx.x * 256 + threadIdx.x;
  if (idx >= 6 * 16384) return;
  int wi = idx >> 14;
  int j  = idx & 16383;
  const float* src;
  switch (wi) {
    case 0: src = wl;  break;
    case 1: src = wr;  break;
    case 2: src = wlg; break;
    case 3: src = wrg; break;
    case 4: src = wog; break;
    default: src = wout; break;
  }
  wbf[idx] = f2bf(src[j]);
}

// ---------------------------------------------------------------------------
// Kernel 1: fused LayerNorm(D) + 5 projections (WMMA) + mask + sigmoid gates.
// One wave handles 16 ij-rows x all 128 h. Outputs channel-major:
//   leftT/rightT: bf16 [h][ij],  ogT: fp32 [h][ij]
// C-tile layout puts each lane's 8 accum values on 8 consecutive ij rows for
// a fixed h => single uint4 (bf16) / two float4 (fp32) stores per lane.
// ---------------------------------------------------------------------------
__global__ void __launch_bounds__(256)
k1_norm_proj_gate(const float* __restrict__ x, const float* __restrict__ mask,
                  const float* __restrict__ nw, const float* __restrict__ nb,
                  const unsigned short* __restrict__ wbf,
                  unsigned short* __restrict__ leftT,
                  unsigned short* __restrict__ rightT,
                  float* __restrict__ ogT) {
  __shared__ float sw[DDIM], sb[DDIM];
  int tid = threadIdx.x;
  if (tid < DDIM) { sw[tid] = nw[tid]; sb[tid] = nb[tid]; }
  __syncthreads();

  const int wid  = tid >> 5;
  const int lane = tid & 31;
  const int half = lane >> 4;     // 0: K 0-15, 1: K 16-31 of each 32-K chunk
  const int mrow = lane & 15;
  const long row0 = ((long)blockIdx.x * 8 + wid) * 16;   // first ij row of tile

  // ---- load this lane's share of the 16x128 x-tile; accumulate stats -------
  float xv[4][16];
  float sum = 0.f, sq = 0.f;
  const float* xr = x + (row0 + mrow) * DDIM;
#pragma unroll
  for (int kt = 0; kt < 4; ++kt) {
    const float4* p = (const float4*)(xr + kt * 32 + half * 16);
    float4 f[4];
    f[0] = p[0]; f[1] = p[1];     // K = base+0..7
    f[2] = p[2]; f[3] = p[3];     // K = base+8..15
    const float* fp = (const float*)f;
#pragma unroll
    for (int e = 0; e < 16; ++e) {
      float v = fp[e];
      xv[kt][e] = v;
      sum += v; sq += v * v;
    }
  }
  // partner lane (xor 16) holds the complementary half of the same row
  sum += __shfl_xor(sum, 16);
  sq  += __shfl_xor(sq, 16);
  const float mu   = sum * (1.0f / DDIM);
  const float var  = sq * (1.0f / DDIM) - mu * mu;
  const float rsig = rsqrtf(var + LN_EPS);

  // ---- build bf16 A fragments of xn (16 rows x 128 K, 4 frags of K=32) -----
  ABFrag a[4];
#pragma unroll
  for (int kt = 0; kt < 4; ++kt) {
#pragma unroll
    for (int e = 0; e < 16; ++e) {
      int K = kt * 32 + half * 16 + e;
      float xn = (xv[kt][e] - mu) * rsig * sw[K] + sb[K];
      a[kt].us[e] = f2bf(xn);
    }
  }

  // ---- mask for this lane's 8 output rows ----------------------------------
  const float4* mp = (const float4*)(mask + row0 + half * 8);
  float4 mf[2]; mf[0] = mp[0]; mf[1] = mp[1];
  const float* mk = (const float*)mf;

  const long ijs = row0 + half * 8;   // base ij of this lane's 8 C values

  // ---- 8 h-tiles x 5 projections, K=128 each -------------------------------
  for (int nt = 0; nt < 8; ++nt) {
    const int h = nt * 16 + mrow;
    floatx8 cl = {0,0,0,0,0,0,0,0}, cr = cl, clg = cl, crg = cl, cog = cl;
#pragma unroll
    for (int kt = 0; kt < 4; ++kt) {
      const int kb = kt * 32 + half * 16;     // B frag: 16 contiguous K per lane
      ABFrag bL, bR, bLG, bRG, bOG;
      const uint4* q;
      q = (const uint4*)(wbf + 0 * 16384 + h * DDIM + kb); bL.q[0]  = q[0]; bL.q[1]  = q[1];
      q = (const uint4*)(wbf + 1 * 16384 + h * DDIM + kb); bR.q[0]  = q[0]; bR.q[1]  = q[1];
      q = (const uint4*)(wbf + 2 * 16384 + h * DDIM + kb); bLG.q[0] = q[0]; bLG.q[1] = q[1];
      q = (const uint4*)(wbf + 3 * 16384 + h * DDIM + kb); bRG.q[0] = q[0]; bRG.q[1] = q[1];
      q = (const uint4*)(wbf + 4 * 16384 + h * DDIM + kb); bOG.q[0] = q[0]; bOG.q[1] = q[1];
      cl  = __builtin_amdgcn_wmma_f32_16x16x32_bf16(false, a[kt].v, false, bL.v,  (short)0, cl,  false, false);
      cr  = __builtin_amdgcn_wmma_f32_16x16x32_bf16(false, a[kt].v, false, bR.v,  (short)0, cr,  false, false);
      clg = __builtin_amdgcn_wmma_f32_16x16x32_bf16(false, a[kt].v, false, bLG.v, (short)0, clg, false, false);
      crg = __builtin_amdgcn_wmma_f32_16x16x32_bf16(false, a[kt].v, false, bRG.v, (short)0, crg, false, false);
      cog = __builtin_amdgcn_wmma_f32_16x16x32_bf16(false, a[kt].v, false, bOG.v, (short)0, cog, false, false);
    }
    // epilogue: mask * sigmoid gates, pack bf16, contiguous stores
    ABFrag lp, rp;
    float og8[8];
#pragma unroll
    for (int v = 0; v < 8; ++v) {
      float m = mk[v];
      lp.us[v] = f2bf(cl[v] * m * sigmoid_f(clg[v]));
      rp.us[v] = f2bf(cr[v] * m * sigmoid_f(crg[v]));
      og8[v]   = sigmoid_f(cog[v]);
    }
    const size_t base = (size_t)h * LL + (size_t)ijs;
    *(uint4*)(leftT + base)  = lp.q[0];
    *(uint4*)(rightT + base) = rp.q[0];
    float4 o0 = {og8[0], og8[1], og8[2], og8[3]};
    float4 o1 = {og8[4], og8[5], og8[6], og8[7]};
    *(float4*)(ogT + base)     = o0;
    *(float4*)(ogT + base + 4) = o1;
  }
}

// ---------------------------------------------------------------------------
// Kernel 2: triangle contraction. For each h: O_h = L_h (512x512) @ R_h^T,
// bf16 inputs / fp32 accum. One wave per 64x64 output tile, 4x4 accum tiles,
// K-loop of 16 (K=32 per WMMA) => 256 WMMAs/wave. left/right are L2-resident.
// ---------------------------------------------------------------------------
__global__ void __launch_bounds__(32)
k2_triangle_gemm(const unsigned short* __restrict__ leftT,
                 const unsigned short* __restrict__ rightT,
                 float* __restrict__ Obuf) {
  const int h  = blockIdx.z;
  const int it = blockIdx.y;
  const int jt = blockIdx.x;
  const int lane = threadIdx.x;
  const int half = lane >> 4;
  const int mrow = lane & 15;

  const unsigned short* Lh = leftT  + (size_t)h * LL;
  const unsigned short* Rh = rightT + (size_t)h * LL;

  floatx8 acc[4][4];
#pragma unroll
  for (int mi = 0; mi < 4; ++mi)
#pragma unroll
    for (int nj = 0; nj < 4; ++nj)
      acc[mi][nj] = (floatx8){0,0,0,0,0,0,0,0};

  for (int kb = 0; kb < 16; ++kb) {
    const int k0 = kb * 32;
    ABFrag a[4], b[4];
#pragma unroll
    for (int mi = 0; mi < 4; ++mi) {
      const uint4* p = (const uint4*)(Lh + (size_t)(it * 64 + mi * 16 + mrow) * LDIM
                                         + k0 + half * 8);
      a[mi].q[0] = p[0];      // K = base+0..7
      a[mi].q[1] = p[2];      // K = base+16..23
    }
#pragma unroll
    for (int nj = 0; nj < 4; ++nj) {
      const uint4* q = (const uint4*)(Rh + (size_t)(jt * 64 + nj * 16 + mrow) * LDIM
                                         + k0 + half * 8);
      b[nj].q[0] = q[0];      // B^T column = R row: K = base+0..7, base+16..23
      b[nj].q[1] = q[2];
    }
#pragma unroll
    for (int mi = 0; mi < 4; ++mi)
#pragma unroll
      for (int nj = 0; nj < 4; ++nj)
        acc[mi][nj] = __builtin_amdgcn_wmma_f32_16x16x32_bf16(
            false, a[mi].v, false, b[nj].v, (short)0, acc[mi][nj], false, false);
  }

  float* Oh = Obuf + (size_t)h * LL;
#pragma unroll
  for (int mi = 0; mi < 4; ++mi)
#pragma unroll
    for (int nj = 0; nj < 4; ++nj)
#pragma unroll
      for (int v = 0; v < 8; ++v) {
        int i = it * 64 + mi * 16 + v + 8 * half;
        int j = jt * 64 + nj * 16 + mrow;
        Oh[(size_t)i * LDIM + j] = acc[mi][nj][v];
      }
}

// ---------------------------------------------------------------------------
// Kernel 3: LayerNorm(H) of O, * og, then WMMA projection with w_out (bf16).
// Block = 256 threads handles a 32-row [ij][h] tile; O/og are transposed from
// channel-major into LDS (pad 129 => conflict-free), stats via shfl tree.
// ---------------------------------------------------------------------------
__global__ void __launch_bounds__(256)
k3_out_norm_proj(const float* __restrict__ Obuf, const float* __restrict__ ogT,
                 const float* __restrict__ onw_g, const float* __restrict__ onb_g,
                 const unsigned short* __restrict__ woutbf,
                 float* __restrict__ y) {
  __shared__ float Ot[32][129];
  __shared__ float Gt[32][129];
  __shared__ float onw[HDIM], onb[HDIM];
  __shared__ float mu_s[32], rs_s[32];

  const int tid = threadIdx.x;
  const long ij0 = (long)blockIdx.x * 32;
  if (tid < HDIM) { onw[tid] = onw_g[tid]; onb[tid] = onb_g[tid]; }

  // stage + transpose: coalesced reads along ij, padded LDS writes
  {
    const int ij = tid & 31;
    const int hb = tid >> 5;
#pragma unroll
    for (int s = 0; s < 16; ++s) {
      int h = s * 8 + hb;
      Ot[ij][h] = Obuf[(size_t)h * LL + ij0 + ij];
      Gt[ij][h] = ogT[(size_t)h * LL + ij0 + ij];
    }
  }
  __syncthreads();

  // per-row stats over 128 h: 8 threads/row, shfl_xor reduce within group
  {
    const int row = tid >> 3;
    const int s   = tid & 7;
    float sum = 0.f, sq = 0.f;
#pragma unroll
    for (int u = 0; u < 16; ++u) {
      float o = Ot[row][s * 16 + u];
      sum += o; sq += o * o;
    }
    sum += __shfl_xor(sum, 1); sq += __shfl_xor(sq, 1);
    sum += __shfl_xor(sum, 2); sq += __shfl_xor(sq, 2);
    sum += __shfl_xor(sum, 4); sq += __shfl_xor(sq, 4);
    if (s == 0) {
      float mu = sum * (1.0f / HDIM);
      mu_s[row] = mu;
      rs_s[row] = rsqrtf(sq * (1.0f / HDIM) - mu * mu + LN_EPS);
    }
  }
  __syncthreads();

  // WMMA: 2 mi-tiles x 8 nt-tiles across 8 waves (2 tiles each), K=128
  const int wavei = tid >> 5;
  const int lane  = tid & 31;
  const int half  = lane >> 4;
  const int mrow  = lane & 15;
  const int mi    = wavei & 1;
  const int ntb   = wavei >> 1;
  const int m     = mi * 16 + mrow;
  const float mu  = mu_s[m];
  const float rs  = rs_s[m];

  ABFrag a[4];
#pragma unroll
  for (int kt = 0; kt < 4; ++kt) {
#pragma unroll
    for (int e = 0; e < 16; ++e) {
      int K = kt * 32 + half * 16 + e;
      float o = Ot[m][K];
      float val = ((o - mu) * rs * onw[K] + onb[K]) * Gt[m][K];
      a[kt].us[e] = f2bf(val);
    }
  }

#pragma unroll
  for (int t = 0; t < 2; ++t) {
    const int nt = ntb + t * 4;
    floatx8 acc = {0,0,0,0,0,0,0,0};
#pragma unroll
    for (int kt = 0; kt < 4; ++kt) {
      ABFrag b;
      const uint4* q = (const uint4*)(woutbf + (nt * 16 + mrow) * HDIM
                                             + kt * 32 + half * 16);
      b.q[0] = q[0]; b.q[1] = q[1];
      acc = __builtin_amdgcn_wmma_f32_16x16x32_bf16(
          false, a[kt].v, false, b.v, (short)0, acc, false, false);
    }
#pragma unroll
    for (int v = 0; v < 8; ++v) {
      long ij = ij0 + mi * 16 + v + 8 * half;
      int  d  = nt * 16 + mrow;
      y[(size_t)ij * DDIM + d] = acc[v];
    }
  }
}

// ---------------------------------------------------------------------------
// Launch: workspace layout (bytes):
//   [0)          wbf     6*16384 bf16            (196,608; slot 262,144)
//   [262,144)    leftT   128*262144 bf16         (67,108,864)
//   [67,371,008) rightT  128*262144 bf16         (67,108,864)
//   [134,479,872) ogT    128*262144 fp32         (134,217,728)
//   [268,697,600) Obuf   128*262144 fp32         (134,217,728)   total ~384 MB
// ---------------------------------------------------------------------------
extern "C" void kernel_launch(void* const* d_in, const int* in_sizes, int n_in,
                              void* d_out, int out_size, void* d_ws, size_t ws_size,
                              hipStream_t stream) {
  (void)in_sizes; (void)n_in; (void)out_size; (void)ws_size;
  const float* x       = (const float*)d_in[0];
  const float* mask    = (const float*)d_in[1];
  const float* norm_w  = (const float*)d_in[2];
  const float* norm_b  = (const float*)d_in[3];
  const float* w_left  = (const float*)d_in[4];
  const float* w_right = (const float*)d_in[5];
  const float* w_lgate = (const float*)d_in[6];
  const float* w_rgate = (const float*)d_in[7];
  const float* w_ogate = (const float*)d_in[8];
  const float* onorm_w = (const float*)d_in[9];
  const float* onorm_b = (const float*)d_in[10];
  const float* w_out   = (const float*)d_in[11];
  float* y = (float*)d_out;

  char* ws = (char*)d_ws;
  unsigned short* wbf    = (unsigned short*)ws;
  unsigned short* leftT  = (unsigned short*)(ws + 262144);
  unsigned short* rightT = (unsigned short*)(ws + 262144 + 67108864);
  float*          ogT    = (float*)(ws + 262144 + 2ull * 67108864);
  float*          Obuf   = (float*)(ws + 262144 + 2ull * 67108864 + 134217728ull);
  unsigned short* woutbf = wbf + 5 * 16384;

  k0_convert_weights<<<384, 256, 0, stream>>>(w_left, w_right, w_lgate, w_rgate,
                                              w_ogate, w_out, wbf);
  k1_norm_proj_gate<<<2048, 256, 0, stream>>>(x, mask, norm_w, norm_b, wbf,
                                              leftT, rightT, ogT);
  k2_triangle_gemm<<<dim3(8, 8, 128), 32, 0, stream>>>(leftT, rightT, Obuf);
  k3_out_norm_proj<<<8192, 256, 0, stream>>>(Obuf, ogT, onorm_w, onorm_b,
                                             woutbf, y);
}
